// HandwritingNetwork_24257975287972
// MI455X (gfx1250) — matrix operand order repaired and checked by reading, not compiled, not run on hardware
//
#include <hip/hip_runtime.h>
#include <hip/hip_bf16.h>
#include <stdint.h>
#include <stddef.h>

// ---------------------------------------------------------------------------
// CDNA5 (gfx1250) WMMA types
// ---------------------------------------------------------------------------
typedef __attribute__((ext_vector_type(16))) __bf16 bf16x16;
typedef __attribute__((ext_vector_type(8)))  float  f32x8;
typedef __attribute__((ext_vector_type(4)))  unsigned int uint4v;

#define WMMA_BF16(A, Bm, C) \
  __builtin_amdgcn_wmma_f32_16x16x32_bf16(false, (A), false, (Bm), (short)0, (C), false, false)

// ---------------------------------------------------------------------------
// Problem constants
// ---------------------------------------------------------------------------
constexpr int B_ = 64, T_ = 400, H_ = 400, ULEN_ = 64;
constexpr int HSZ = B_ * H_;            // 25600 floats (one c-state buffer)
constexpr int NT_LSTM = 100;            // 1600/16 column tiles
constexpr int KT_L1 = 15;               // 467 -> 480
constexpr int KT_L23 = 28;              // 867 -> 896
constexpr int KT_OUT = 13;              // 400 -> 416
constexpr int BT_ = B_ * T_;            // 25600 rows

// packed bf16 activation layouts (combined-input row orderings, x at tail)
constexpr int LD1  = 480;   // IN1/INA: [w(0..63), h1(64..463), x(464..466), pad]
constexpr int LD23 = 896;   // IN2/IN3: [h_below(0..399), w(400..463), h_prev(464..863), x(864..866), pad]
constexpr int LDO  = 416;   // OUTS:    [h3(0..399), pad]
constexpr int S1_W = 0,  S1_H1 = 64,  S1_X = 464;
constexpr int S2_HB = 0, S2_W = 400, S2_HP = 464, S2_X = 864;

// workspace layout (bytes)
constexpr size_t SZ_TILE = 1024;
constexpr size_t OFF_WB1 = 0;
constexpr size_t SZ_WB1  = (size_t)KT_L1  * NT_LSTM * SZ_TILE;
constexpr size_t OFF_WB2 = OFF_WB1 + SZ_WB1;
constexpr size_t SZ_WB23 = (size_t)KT_L23 * NT_LSTM * SZ_TILE;
constexpr size_t OFF_WB3 = OFF_WB2 + SZ_WB23;
constexpr size_t OFF_WBA = OFF_WB3 + SZ_WB23;
constexpr size_t SZ_WBA  = (size_t)KT_L1 * 2 * SZ_TILE;
constexpr size_t OFF_WBO = OFF_WBA + SZ_WBA;
constexpr size_t SZ_WBO  = (size_t)KT_OUT * 8 * SZ_TILE;

constexpr size_t OFF_ACT = OFF_WBO + SZ_WBO;        // bf16 region (ushort counts)
constexpr size_t U_IN1  = (size_t)2 * B_ * LD1;     // ping-pong
constexpr size_t U_INA  = (size_t)2 * B_ * LD1;
constexpr size_t U_IN2  = (size_t)2 * B_ * LD23;
constexpr size_t U_IN3  = (size_t)2 * B_ * LD23;
constexpr size_t U_OUTS = (size_t)BT_ * LDO;
constexpr size_t U_TOTAL = U_IN1 + U_INA + U_IN2 + U_IN3 + U_OUTS;
constexpr size_t OFF_F32 = OFF_ACT + 2 * U_TOTAL;   // f32 region: c states + kappa

// ---------------------------------------------------------------------------
// helpers
// ---------------------------------------------------------------------------
__device__ __forceinline__ unsigned short f2bf(float f) {
  union { float f; unsigned u; } cv; cv.f = f;
  unsigned u = cv.u;
  return (unsigned short)((u + 0x7fffu + ((u >> 16) & 1u)) >> 16); // RNE
}

__device__ __forceinline__ float sigf(float x) { return 1.f / (1.f + __expf(-x)); }

union AFrag { bf16x16 v; uint4v q[2]; unsigned short h[16]; };

// A fragment: lane supplies row b, k-pairs {half*8+0..7} and {half*8+16..23}
// within the k-tile -> two contiguous 16B chunks of the packed bf16 input row.
__device__ __forceinline__ void loadAfrag(AFrag& a, const unsigned short* __restrict__ rowp,
                                          int kt) {
  const unsigned short* p = rowp + kt * 32;
  a.q[0] = *reinterpret_cast<const uint4v*>(p);
  a.q[1] = *reinterpret_cast<const uint4v*>(p + 16);
}

// ---------------------------------------------------------------------------
// Weight swizzle with row permutation (up to 4 source segments) ->
// bf16 B-fragment tiles. B tile (32x16): n = nt*16+(lane&15),
// element s holds k = kt*32 + (lane>>4)*8 + (s&7) + (s>>3)*16.
// ---------------------------------------------------------------------------
__global__ void swizzleB_kernel(const float* __restrict__ s0, int o0, int c0,
                                const float* __restrict__ s1, int o1, int c1,
                                const float* __restrict__ s2, int o2, int c2,
                                const float* __restrict__ s3, int o3, int c3,
                                int Nact, int NT, int KT,
                                unsigned short* __restrict__ dst) {
  int tid  = blockIdx.x * blockDim.x + threadIdx.x;
  int lane = tid & 31;
  int tile = tid >> 5;
  if (tile >= KT * NT) return;
  int kt = tile / NT, nt = tile % NT;
  int n = nt * 16 + (lane & 15);
  int half = lane >> 4;
  unsigned short vals[16];
#pragma unroll
  for (int v = 0; v < 8; ++v) {
#pragma unroll
    for (int p = 0; p < 2; ++p) {
      int k = kt * 32 + half * 8 + ((2 * v + p) & 7) + ((2 * v + p) >> 3) * 16;
      float f = 0.f;
      if (n < Nact) {
        const float* s = nullptr; int row = 0; int rem = k;
        if (rem < c0) { s = s0; row = o0 + rem; }
        else if ((rem -= c0) < c1) { s = s1; row = o1 + rem; }
        else if ((rem -= c1) < c2) { s = s2; row = o2 + rem; }
        else if ((rem -= c2) < c3) { s = s3; row = o3 + rem; }
        if (s) f = s[(size_t)row * Nact + n];
      }
      vals[2 * v + p] = f2bf(f);
    }
  }
  unsigned short* o = dst + ((size_t)tile * 32 + lane) * 16;
#pragma unroll
  for (int i = 0; i < 16; ++i) o[i] = vals[i];
}

__global__ void zero_u32_kernel(unsigned int* __restrict__ p, size_t n) {
  size_t i = (size_t)blockIdx.x * blockDim.x + threadIdx.x;
  if (i < n) p[i] = 0u;
}
__global__ void zero_f32_kernel(float* __restrict__ p, int n) {
  int i = blockIdx.x * blockDim.x + threadIdx.x;
  if (i < n) p[i] = 0.f;
}

// seed x_t (bf16) into the four x-slots of the step-t input buffers
__global__ void seedx_kernel(const float* __restrict__ x, int t,
                             unsigned short* __restrict__ xs0,  // IN1 x-slot, ld LD1
                             unsigned short* __restrict__ xs1,  // INA x-slot, ld LD1
                             unsigned short* __restrict__ xs2,  // IN2 x-slot, ld LD23
                             unsigned short* __restrict__ xs3)  // IN3 x-slot, ld LD23
{
  int i = blockIdx.x * blockDim.x + threadIdx.x;
  if (i < 192) {
    int b = i / 3, d = i % 3;
    unsigned short xv = f2bf(x[((size_t)b * T_ + t) * 3 + d]);
    xs0[(size_t)b * LD1 + d] = xv;
    xs1[(size_t)b * LD1 + d] = xv;
    xs2[(size_t)b * LD23 + d] = xv;
    xs3[(size_t)b * LD23 + d] = xv;
  }
}

// ---------------------------------------------------------------------------
// One LSTM layer step. grid = 25 blocks (hidden tiles), block = 128 (4 waves).
// Wave: one 16-batch M tile x the 4 gate column tiles of its hidden tile.
// Epilogue fuses gates, masked carry, and bf16 scatter to consumer slots.
// ---------------------------------------------------------------------------
template<int KT>
__global__ __launch_bounds__(128) void lstm_kernel(
    const unsigned short* __restrict__ inb, int ldin,   // packed bf16 [64][ldin]
    const unsigned short* __restrict__ wsB,             // swizzled weights
    const float* __restrict__ bias,                     // [1600]
    const float* __restrict__ c_rd, float* __restrict__ c_wr,
    const int* __restrict__ x_len, int t,
    const unsigned short* __restrict__ hold, int ldh,   // old-h slot (read buf)
    unsigned short* __restrict__ d0, int ld0,           // h destinations (bf16)
    unsigned short* __restrict__ d1, int ld1,
    unsigned short* __restrict__ d2, int ld2,
    unsigned short* __restrict__ outs)                  // layer3: packed trace
{
  const int lane = threadIdx.x & 31;
  const int wid  = threadIdx.x >> 5;
  const int half = lane >> 4;
  const int mrow = lane & 15;
  const int jt = blockIdx.x;
  const int ab = wid * 16 + mrow;

  const bf16x16* __restrict__ bt = reinterpret_cast<const bf16x16*>(wsB);
  const unsigned short* __restrict__ rowp = inb + (size_t)ab * ldin + half * 8;

  f32x8 acc0 = {}, acc1 = {}, acc2 = {}, acc3 = {};
  for (int kt = 0; kt < KT; ++kt) {
    AFrag a;
    loadAfrag(a, rowp, kt);
    bf16x16 b0 = bt[(size_t)(kt * NT_LSTM + jt +  0) * 32 + lane];
    bf16x16 b1 = bt[(size_t)(kt * NT_LSTM + jt + 25) * 32 + lane];
    bf16x16 b2 = bt[(size_t)(kt * NT_LSTM + jt + 50) * 32 + lane];
    bf16x16 b3 = bt[(size_t)(kt * NT_LSTM + jt + 75) * 32 + lane];
    acc0 = WMMA_BF16(a.v, b0, acc0);
    acc1 = WMMA_BF16(a.v, b1, acc1);
    acc2 = WMMA_BF16(a.v, b2, acc2);
    acc3 = WMMA_BF16(a.v, b3, acc3);
  }

  const int j = jt * 16 + mrow;           // C/D: m = 8*half + r, n = mrow
  const float bi  = bias[j];
  const float bfv = bias[H_ + j];
  const float bg  = bias[2 * H_ + j];
  const float bov = bias[3 * H_ + j];
#pragma unroll
  for (int r = 0; r < 8; ++r) {
    int b = wid * 16 + half * 8 + r;
    float zi = acc0[r] + bi;
    float zf = acc1[r] + bfv;
    float zg = acc2[r] + bg;
    float zo = acc3[r] + bov;
    float co = c_rd[b * H_ + j];
    float cn = sigf(zf) * co + sigf(zi) * tanhf(zg);
    float hn = sigf(zo) * tanhf(cn);
    bool valid = t < x_len[b];
    c_wr[b * H_ + j] = valid ? cn : co;
    unsigned short hb = valid ? f2bf(hn) : hold[(size_t)b * ldh + j];
    if (d0) d0[(size_t)b * ld0 + j] = hb;
    if (d1) d1[(size_t)b * ld1 + j] = hb;
    if (d2) d2[(size_t)b * ld2 + j] = hb;
    if (outs) outs[((size_t)b * T_ + t) * LDO + j] = valid ? f2bf(hn) : (unsigned short)0;
  }
}

// ---------------------------------------------------------------------------
// Attention window (single workgroup): ap = INA @ Wa + ba via WMMA, then
// GMM phi + one-hot gather; writes blended bf16 window to 4 consumer slots
// and seeds x_{t+1} into next-step buffers.
// ---------------------------------------------------------------------------
__global__ __launch_bounds__(128) void attn_kernel(
    const unsigned short* __restrict__ ina,             // INA[rd], ld LD1
    const unsigned short* __restrict__ wsBa, const float* __restrict__ ba,
    const int* __restrict__ x_len, int t,
    const int* __restrict__ c, const int* __restrict__ c_len,
    const float* __restrict__ kap_rd, float* __restrict__ kap_wr,
    unsigned short* __restrict__ wd0,   // IN2[rd]+S2_W  (ld LD23)
    unsigned short* __restrict__ wd1,   // IN3[rd]+S2_W  (ld LD23)
    unsigned short* __restrict__ wd2,   // IN1[wr]+S1_W  (ld LD1)
    unsigned short* __restrict__ wd3,   // INA[wr]+S1_W  (ld LD1)
    const float* __restrict__ x,
    unsigned short* __restrict__ xs0,   // IN1[wr]+S1_X (ld LD1)
    unsigned short* __restrict__ xs1,   // INA[wr]+S1_X (ld LD1)
    unsigned short* __restrict__ xs2,   // IN2[wr]+S2_X (ld LD23)
    unsigned short* __restrict__ xs3)   // IN3[wr]+S2_X (ld LD23)
{
  __shared__ float ap_s[64][33];
  __shared__ float wn_s[64][65];
  const int lane = threadIdx.x & 31;
  const int wid  = threadIdx.x >> 5;
  const int half = lane >> 4;
  const int mrow = lane & 15;
  const int ab = wid * 16 + mrow;

  const bf16x16* __restrict__ bt = reinterpret_cast<const bf16x16*>(wsBa);
  const unsigned short* __restrict__ rowp = ina + (size_t)ab * LD1 + half * 8;

  f32x8 acc0 = {}, acc1 = {};
  for (int kt = 0; kt < KT_L1; ++kt) {
    AFrag a;
    loadAfrag(a, rowp, kt);
    bf16x16 b0 = bt[(size_t)(kt * 2 + 0) * 32 + lane];
    bf16x16 b1 = bt[(size_t)(kt * 2 + 1) * 32 + lane];
    acc0 = WMMA_BF16(a.v, b0, acc0);
    acc1 = WMMA_BF16(a.v, b1, acc1);
  }
#pragma unroll
  for (int r = 0; r < 8; ++r) {
    int m = wid * 16 + half * 8 + r;
    int n0 = mrow, n1 = 16 + mrow;
    ap_s[m][n0] = acc0[r] + ((n0 < 30) ? ba[n0] : 0.f);
    ap_s[m][n1] = acc1[r] + ((n1 < 30) ? ba[n1] : 0.f);
  }
  __syncthreads();

  const int tid = threadIdx.x;
  if (tid < 64) {
    const int b = tid;
    float kap[10], ea[10], eb[10];
#pragma unroll
    for (int k = 0; k < 10; ++k) {
      kap[k] = kap_rd[b * 10 + k] + __expf(ap_s[b][20 + k]) * 0.04f;  // +exp(k_hat)/25
      ea[k]  = __expf(ap_s[b][k]);
      eb[k]  = __expf(ap_s[b][10 + k]);
    }
    for (int v = 0; v < 64; ++v) wn_s[b][v] = 0.f;
    const int cl = c_len[b];
    for (int u = 0; u < ULEN_; ++u) {
      if (u < cl) {
        float uf = (float)u, phi = 0.f;
#pragma unroll
        for (int k = 0; k < 10; ++k) {
          float d = kap[k] - uf;
          phi += ea[k] * __expf(-eb[k] * d * d);
        }
        wn_s[b][c[b * ULEN_ + u]] += phi;   // einsum vs one-hot == gather-add
      }
    }
    const bool valid = t < x_len[b];
#pragma unroll
    for (int k = 0; k < 10; ++k)
      kap_wr[b * 10 + k] = valid ? kap[k] : kap_rd[b * 10 + k];
    for (int v = 0; v < 64; ++v) {
      unsigned short wb = valid ? f2bf(wn_s[b][v])
                                : ina[(size_t)b * LD1 + S1_W + v];
      wd0[(size_t)b * LD23 + v] = wb;
      wd1[(size_t)b * LD23 + v] = wb;
      wd2[(size_t)b * LD1  + v] = wb;
      wd3[(size_t)b * LD1  + v] = wb;
    }
  }
  // seed x_{t+1} for next step's input buffers
  if (t + 1 < T_) {
    for (int i = tid; i < 192; i += 128) {
      int b = i / 3, d = i % 3;
      unsigned short xv = f2bf(x[((size_t)b * T_ + (t + 1)) * 3 + d]);
      xs0[(size_t)b * LD1 + d] = xv;
      xs1[(size_t)b * LD1 + d] = xv;
      xs2[(size_t)b * LD23 + d] = xv;
      xs3[(size_t)b * LD23 + d] = xv;
    }
  }
}

// ---------------------------------------------------------------------------
// Final projection: [B*T,400] @ [400,121] + bo from the packed bf16 trace.
// ---------------------------------------------------------------------------
__global__ __launch_bounds__(128) void outproj_kernel(
    const unsigned short* __restrict__ outs,
    const unsigned short* __restrict__ wsBo,
    const float* __restrict__ bo,
    float* __restrict__ out) {
  const int lane = threadIdx.x & 31;
  const int wid  = threadIdx.x >> 5;
  const int half = lane >> 4;
  const int mrow = lane & 15;
  const int mt = blockIdx.x * 4 + wid;      // 0..1599
  const int arow = mt * 16 + mrow;

  const bf16x16* __restrict__ bt = reinterpret_cast<const bf16x16*>(wsBo);
  const unsigned short* __restrict__ rowp = outs + (size_t)arow * LDO + half * 8;

  f32x8 acc[8] = {};
  for (int kt = 0; kt < KT_OUT; ++kt) {
    AFrag a;
    loadAfrag(a, rowp, kt);
#pragma unroll
    for (int nt = 0; nt < 8; ++nt) {
      bf16x16 bm = bt[(size_t)(kt * 8 + nt) * 32 + lane];
      acc[nt] = WMMA_BF16(a.v, bm, acc[nt]);
    }
  }
#pragma unroll
  for (int nt = 0; nt < 8; ++nt) {
    int ncol = nt * 16 + mrow;
    if (ncol < 121) {
      float bb = bo[ncol];
#pragma unroll
      for (int r = 0; r < 8; ++r) {
        int row = mt * 16 + half * 8 + r;
        out[(size_t)row * 121 + ncol] = acc[nt][r] + bb;
      }
    }
  }
}

// ---------------------------------------------------------------------------
// launch
// ---------------------------------------------------------------------------
extern "C" void kernel_launch(void* const* d_in, const int* in_sizes, int n_in,
                              void* d_out, int out_size, void* d_ws, size_t ws_size,
                              hipStream_t stream) {
  (void)in_sizes; (void)n_in; (void)out_size; (void)ws_size;
  const float* x     = (const float*)d_in[0];
  const int*   x_len = (const int*)d_in[1];
  const int*   c     = (const int*)d_in[2];
  const int*   c_len = (const int*)d_in[3];
  // d_in[4] = bias: unused by the reference computation
  const float* W1 = (const float*)d_in[5];
  const float* U1 = (const float*)d_in[6];
  const float* b1 = (const float*)d_in[7];
  const float* Wa = (const float*)d_in[8];
  const float* ba = (const float*)d_in[9];
  const float* W2 = (const float*)d_in[10];
  const float* U2 = (const float*)d_in[11];
  const float* b2 = (const float*)d_in[12];
  const float* W3 = (const float*)d_in[13];
  const float* U3 = (const float*)d_in[14];
  const float* b3 = (const float*)d_in[15];
  const float* Wo = (const float*)d_in[16];
  const float* bo = (const float*)d_in[17];
  float* out = (float*)d_out;

  char* ws = (char*)d_ws;
  unsigned short* WB1 = (unsigned short*)(ws + OFF_WB1);
  unsigned short* WB2 = (unsigned short*)(ws + OFF_WB2);
  unsigned short* WB3 = (unsigned short*)(ws + OFF_WB3);
  unsigned short* WBA = (unsigned short*)(ws + OFF_WBA);
  unsigned short* WBO = (unsigned short*)(ws + OFF_WBO);

  unsigned short* act = (unsigned short*)(ws + OFF_ACT);
  unsigned short* IN1  = act;                   // [2][64][LD1]
  unsigned short* INA  = IN1 + U_IN1;           // [2][64][LD1]
  unsigned short* IN2  = INA + U_INA;           // [2][64][LD23]
  unsigned short* IN3  = IN2 + U_IN2;           // [2][64][LD23]
  unsigned short* OUTS = IN3 + U_IN3;           // [BT][LDO]

  float* fst = (float*)(ws + OFF_F32);
  float* C1  = fst;                 // [2][HSZ]
  float* C2  = C1 + 2 * HSZ;
  float* C3  = C2 + 2 * HSZ;
  float* KAP = C3 + 2 * HSZ;        // [2][640]

  const size_t PB1 = (size_t)B_ * LD1;   // one ping buffer (ushorts)
  const size_t PB23 = (size_t)B_ * LD23;

  // 1) weight swizzle (with row permutation so x sits at the tail)
  swizzleB_kernel<<<(KT_L1 * NT_LSTM * 32 + 127) / 128, 128, 0, stream>>>(
      W1, 0, 64,  U1, 0, 400,  W1, 64, 3,  nullptr, 0, 0, 1600, NT_LSTM, KT_L1, WB1);
  swizzleB_kernel<<<(KT_L23 * NT_LSTM * 32 + 127) / 128, 128, 0, stream>>>(
      W2, 3, 400, W2, 403, 64, U2, 0, 400, W2, 0, 3, 1600, NT_LSTM, KT_L23, WB2);
  swizzleB_kernel<<<(KT_L23 * NT_LSTM * 32 + 127) / 128, 128, 0, stream>>>(
      W3, 3, 400, W3, 403, 64, U3, 0, 400, W3, 0, 3, 1600, NT_LSTM, KT_L23, WB3);
  swizzleB_kernel<<<(KT_L1 * 2 * 32 + 127) / 128, 128, 0, stream>>>(
      Wa, 0, 64,  Wa, 67, 400, Wa, 64, 3,  nullptr, 0, 0, 30, 2, KT_L1, WBA);
  swizzleB_kernel<<<(KT_OUT * 8 * 32 + 127) / 128, 128, 0, stream>>>(
      Wo, 0, 400, nullptr, 0, 0, nullptr, 0, 0, nullptr, 0, 0, 121, 8, KT_OUT, WBO);

  // 2) zero packed activations (incl. pads) + f32 states, seed x_0
  {
    size_t n32 = U_TOTAL / 2;
    zero_u32_kernel<<<(unsigned)((n32 + 255) / 256), 256, 0, stream>>>((unsigned int*)act, n32);
    int nf = 6 * HSZ + 2 * 640;
    zero_f32_kernel<<<(nf + 255) / 256, 256, 0, stream>>>(fst, nf);
    seedx_kernel<<<1, 192, 0, stream>>>(x, 0,
        IN1 + S1_X, INA + S1_X, IN2 + S2_X, IN3 + S2_X);
  }

  // 3) sequential scan; stream order is the grid-wide barrier
  for (int t = 0; t < T_; ++t) {
    const int rd = t & 1, wr = rd ^ 1;
    unsigned short* in1r = IN1 + rd * PB1;  unsigned short* in1w = IN1 + wr * PB1;
    unsigned short* inar = INA + rd * PB1;  unsigned short* inaw = INA + wr * PB1;
    unsigned short* in2r = IN2 + rd * PB23; unsigned short* in2w = IN2 + wr * PB23;
    unsigned short* in3r = IN3 + rd * PB23; unsigned short* in3w = IN3 + wr * PB23;

    lstm_kernel<KT_L1><<<25, 128, 0, stream>>>(
        in1r, LD1, WB1, b1, C1 + rd * HSZ, C1 + wr * HSZ, x_len, t,
        in1r + S1_H1, LD1,
        inar + S1_H1, LD1,      // attn input (h1n)
        in2r + S2_HB, LD23,     // lstm2 input (h1n)
        in1w + S1_H1, LD1,      // next-step lstm1 input (carry)
        nullptr);

    attn_kernel<<<1, 128, 0, stream>>>(
        inar, WBA, ba, x_len, t, c, c_len,
        KAP + rd * 640, KAP + wr * 640,
        in2r + S2_W, in3r + S2_W, in1w + S1_W, inaw + S1_W,
        x, in1w + S1_X, inaw + S1_X, in2w + S2_X, in3w + S2_X);

    lstm_kernel<KT_L23><<<25, 128, 0, stream>>>(
        in2r, LD23, WB2, b2, C2 + rd * HSZ, C2 + wr * HSZ, x_len, t,
        in2r + S2_HP, LD23,
        in3r + S2_HB, LD23,     // lstm3 input (h2n)
        in2w + S2_HP, LD23,     // next-step carry
        nullptr, 0,
        nullptr);

    lstm_kernel<KT_L23><<<25, 128, 0, stream>>>(
        in3r, LD23, WB3, b3, C3 + rd * HSZ, C3 + wr * HSZ, x_len, t,
        in3r + S2_HP, LD23,
        in3w + S2_HP, LD23,     // next-step carry
        nullptr, 0,
        nullptr, 0,
        OUTS);

    (void)in1w;
  }

  // 4) batched output projection
  outproj_kernel<<<BT_ / 64, 128, 0, stream>>>(OUTS, WBO, bo, out);
}
